// RelationalLayer_19344532701280
// MI455X (gfx1250) — matrix operand order repaired and checked by reading, not compiled, not run on hardware
//
#include <hip/hip_runtime.h>
#include <math.h>

// ---------- types ----------
typedef float v8f      __attribute__((ext_vector_type(8)));
typedef __bf16 bf16x16 __attribute__((ext_vector_type(16)));
typedef __bf16 bf16x8  __attribute__((ext_vector_type(8)));

union AFrag { bf16x16 v; bf16x8 h[2]; };

// ---------- problem sizes ----------
#define B_   32
#define D_   64
#define KF   14
#define Q_   256
#define G0   511
#define G0P  512     // padded K of big GEMM
#define G1   512

// ---------- workspace layout (in floats) ----------
#define AF_OFF   0
#define CF_OFF   (B_*D_*G0P)                 // 1048576
#define QADD_OFF (2*B_*D_*G0P)               // 2097152
#define XG_OFF   (QADD_OFF + B_*G1)          // 2113536
#define BT_OFF_F (XG_OFF + B_*G1)            // 2129920 ; Bt = bf16[512*512] after this

// =====================================================================
// Kernel 1: af[b,q,k] = x[b,q,:]@g0_w[:14,k] + g0_b[k]
//           cf[b,p,k] = x[b,p,:]@g0_w[14:,k]        (k=511 padded to 0)
// =====================================================================
__global__ void proj_g0(const float* __restrict__ x, const float* __restrict__ g0w,
                        const float* __restrict__ g0b,
                        float* __restrict__ af, float* __restrict__ cf) {
    const int bq = blockIdx.x;          // 0 .. B*D-1
    const int t  = threadIdx.x;         // 256
    __shared__ float xv[KF];
    if (t < KF) xv[t] = x[bq * KF + t];
    __syncthreads();
    for (int k = t; k < G0P; k += 256) {
        float a = 0.f, c = 0.f;
        if (k < G0) {
            a = g0b[k];
#pragma unroll
            for (int j = 0; j < KF; ++j) {
                a += xv[j] * g0w[j * G0 + k];
                c += xv[j] * g0w[(KF + j) * G0 + k];
            }
        }
        af[(size_t)bq * G0P + k] = a;
        cf[(size_t)bq * G0P + k] = c;
    }
}

// =====================================================================
// Kernel 2: qadd[b,n] = qst[b,:]@g1_w[511:,n] + g1_b[n]
// =====================================================================
__global__ void proj_q(const float* __restrict__ qst, const float* __restrict__ g1w,
                       const float* __restrict__ g1b, float* __restrict__ qadd) {
    const int b = blockIdx.x, t = threadIdx.x;   // 256 threads
    __shared__ float qv[Q_];
    qv[t] = qst[b * Q_ + t];
    __syncthreads();
    for (int n = t; n < G1; n += 256) {
        float acc = g1b[n];
        for (int j = 0; j < Q_; ++j) acc += qv[j] * g1w[(G0 + j) * G1 + n];
        qadd[b * G1 + n] = acc;
    }
}

// =====================================================================
// Kernel 3: Bt[n][k] = bf16(g1_w[k][n]) (k<511, else 0)  and zero xg
// Column-major-per-lane layout => B fragment = 32 contiguous bytes/lane.
// =====================================================================
__global__ void prep_bt_xg(const float* __restrict__ g1w, __bf16* __restrict__ Bt,
                           float* __restrict__ xg) {
    const int idx = blockIdx.x * 256 + threadIdx.x;  // 0 .. 262143
    const int n = idx >> 9, k = idx & 511;
    Bt[(size_t)n * 512 + k] = (k < G0) ? (__bf16)g1w[(size_t)k * G1 + n] : (__bf16)0.f;
    if (idx < B_ * G1) xg[idx] = 0.f;
}

// =====================================================================
// Kernel 4: main WMMA GEMM. One block per (p, b).
//   A-tile (64 q-rows x 512 k) = relu(af[b,q,:]+cf[b,p,:]) in bf16,
//   staged in LDS pre-swizzled to the A-fragment layout, contiguous k order (k = 16h + s).
//   8 waves x (64 rows x 64 cols) accumulators; K-loop of 16 x WMMA k32.
//   Epilogue: relu(acc + qadd) -> row-sum -> atomicAdd into xg[b,:].
// =====================================================================
__global__ void __launch_bounds__(256)
rn_gemm(const float* __restrict__ af, const float* __restrict__ cf,
        const __bf16* __restrict__ Bt, const float* __restrict__ qadd,
        float* __restrict__ xg) {
    const int p = blockIdx.x, b = blockIdx.y;
    const int t = threadIdx.x;
    __shared__ __align__(16) unsigned char lds[65536];  // 64 blocks x 1KB

    // ---- stage A tile: 4 threads per q-row, 128 k each, chunks of 8 ----
    {
        const int q = t >> 2, quarter = t & 3;
        const float4* ap = (const float4*)(af + (size_t)(b * D_ + q) * G0P);
        const float4* cp = (const float4*)(cf + (size_t)(b * D_ + p) * G0P);
#pragma unroll
        for (int ch = 0; ch < 16; ++ch) {
            const int k0 = quarter * 128 + ch * 8;
            float4 A0 = ap[k0 / 4], A1 = ap[k0 / 4 + 1];
            float4 C0 = cp[k0 / 4], C1 = cp[k0 / 4 + 1];
            bf16x8 r;
            r[0] = (__bf16)fmaxf(A0.x + C0.x, 0.f);
            r[1] = (__bf16)fmaxf(A0.y + C0.y, 0.f);
            r[2] = (__bf16)fmaxf(A0.z + C0.z, 0.f);
            r[3] = (__bf16)fmaxf(A0.w + C0.w, 0.f);
            r[4] = (__bf16)fmaxf(A1.x + C1.x, 0.f);
            r[5] = (__bf16)fmaxf(A1.y + C1.y, 0.f);
            r[6] = (__bf16)fmaxf(A1.z + C1.z, 0.f);
            r[7] = (__bf16)fmaxf(A1.w + C1.w, 0.f);
            // fragment placement: block (mt,kb); lane/half for contiguous k = 16h + s
            const int mt = q >> 4, kb = k0 >> 5, kk0 = k0 & 31;
            const int lane_s = (q & 15) + (((kk0 >> 4) & 1) << 4);
            const int half   = (kk0 >> 3) & 1;
            const int addr = ((mt * 16 + kb) << 10) + (half << 9) + (lane_s << 4);
            *(bf16x8*)(lds + addr) = r;
        }
    }
    __syncthreads();

    const int wave = t >> 5, lane = t & 31;
    const int wcol = wave * 64;                 // this wave's N base (full N covered by 8 waves)

    v8f acc[4][4] = {};
#pragma unroll 1
    for (int kb = 0; kb < 16; ++kb) {
        AFrag afr[4];
#pragma unroll
        for (int mt = 0; mt < 4; ++mt) {
            const int base = ((mt * 16 + kb) << 10);
            afr[mt].h[0] = *(const bf16x8*)(lds + base + lane * 16);
            afr[mt].h[1] = *(const bf16x8*)(lds + base + 512 + lane * 16);
        }
        bf16x16 bfr[4];
#pragma unroll
        for (int nt = 0; nt < 4; ++nt) {
            const __bf16* bp = Bt + (size_t)(wcol + nt * 16 + (lane & 15)) * 512
                                  + kb * 32 + (lane >> 4) * 16;
            bfr[nt] = *(const bf16x16*)bp;
        }
#pragma unroll
        for (int mt = 0; mt < 4; ++mt)
#pragma unroll
            for (int nt = 0; nt < 4; ++nt)
                acc[mt][nt] = __builtin_amdgcn_wmma_f32_16x16x32_bf16(
                    false, afr[mt].v, false, bfr[nt], (short)0, acc[mt][nt],
                    false, false);
    }

    // ---- epilogue: relu(hq + qproj + g1_b), sum over 64 q-rows, atomic into xg ----
#pragma unroll
    for (int nt = 0; nt < 4; ++nt) {
        const int n = wcol + nt * 16 + (lane & 15);
        const float qv = qadd[b * G1 + n];
        float s = 0.f;
#pragma unroll
        for (int mt = 0; mt < 4; ++mt)
#pragma unroll
            for (int r = 0; r < 8; ++r)
                s += fmaxf(acc[mt][nt][r] + qv, 0.f);
        s += __shfl_xor(s, 16, 32);             // combine the two lane-halves (rows 0-7 vs 8-15 sets)
        if (lane < 16) atomicAdd(&xg[b * G1 + n], s);
    }
}

// =====================================================================
// Kernel 5: f-MLP + log_softmax (tiny). One block per batch row.
// =====================================================================
__global__ void mlp_head(const float* __restrict__ xg,
                         const float* __restrict__ f1w, const float* __restrict__ f1b,
                         const float* __restrict__ f2w, const float* __restrict__ f2b,
                         const float* __restrict__ f3w, const float* __restrict__ f3b,
                         float* __restrict__ out) {
    const int b = blockIdx.x, t = threadIdx.x;   // 256 threads
    __shared__ float xr[512];
    __shared__ float h1[512];
    __shared__ float h2[1024];
    __shared__ float lg[16];
    xr[t] = xg[b * 512 + t];
    xr[t + 256] = xg[b * 512 + t + 256];
    __syncthreads();
    for (int o = t; o < 512; o += 256) {
        float acc = f1b[o];
        for (int i = 0; i < 512; ++i) acc += xr[i] * f1w[(size_t)i * 512 + o];
        h1[o] = fmaxf(acc, 0.f);
    }
    __syncthreads();
    for (int o = t; o < 1024; o += 256) {
        float acc = f2b[o];
        for (int i = 0; i < 512; ++i) acc += h1[i] * f2w[(size_t)i * 1024 + o];
        h2[o] = fmaxf(acc, 0.f);
    }
    __syncthreads();
    if (t < 16) {
        float acc = f3b[t];
        for (int i = 0; i < 1024; ++i) acc += h2[i] * f3w[i * 16 + t];
        lg[t] = acc;
    }
    __syncthreads();
    if (t == 0) {
        float m = lg[0];
        for (int o = 1; o < 16; ++o) m = fmaxf(m, lg[o]);
        float se = 0.f;
        for (int o = 0; o < 16; ++o) se += __expf(lg[o] - m);
        const float lse = m + __logf(se);
        for (int o = 0; o < 16; ++o) out[b * 16 + o] = lg[o] - lse;
    }
}

// =====================================================================
extern "C" void kernel_launch(void* const* d_in, const int* in_sizes, int n_in,
                              void* d_out, int out_size, void* d_ws, size_t ws_size,
                              hipStream_t stream) {
    const float* x   = (const float*)d_in[0];
    const float* qst = (const float*)d_in[1];
    const float* g0w = (const float*)d_in[2];
    const float* g0b = (const float*)d_in[3];
    const float* g1w = (const float*)d_in[4];
    const float* g1b = (const float*)d_in[5];
    const float* f1w = (const float*)d_in[6];
    const float* f1b = (const float*)d_in[7];
    const float* f2w = (const float*)d_in[8];
    const float* f2b = (const float*)d_in[9];
    const float* f3w = (const float*)d_in[10];
    const float* f3b = (const float*)d_in[11];
    float* out = (float*)d_out;

    float* ws   = (float*)d_ws;
    float* af   = ws + AF_OFF;
    float* cf   = ws + CF_OFF;
    float* qadd = ws + QADD_OFF;
    float* xg   = ws + XG_OFF;
    __bf16* Bt  = (__bf16*)(ws + BT_OFF_F);

    proj_g0  <<<dim3(B_ * D_), 256, 0, stream>>>(x, g0w, g0b, af, cf);
    proj_q   <<<dim3(B_),      256, 0, stream>>>(qst, g1w, g1b, qadd);
    prep_bt_xg<<<dim3(1024),   256, 0, stream>>>(g1w, Bt, xg);
    rn_gemm  <<<dim3(D_, B_),  256, 0, stream>>>(af, cf, Bt, qadd, xg);
    mlp_head <<<dim3(B_),      256, 0, stream>>>(xg, f1w, f1b, f2w, f2b, f3w, f3b, out);
}
